// MultiHeadAttention_37778532336275
// MI455X (gfx1250) — matrix operand order repaired and checked by reading, not compiled, not run on hardware
//
#include <hip/hip_runtime.h>
#include <hip/hip_bf16.h>

// Problem constants (match reference)
#define BB 4
#define SS 2048
#define EE 1024
#define HH 16
#define DD 64
#define MM (BB * SS)   // 8192 rows
#define LN_EPS 1e-5f

typedef __attribute__((ext_vector_type(16))) __bf16         bf16x16;
typedef __attribute__((ext_vector_type(8)))  float          f32x8;
typedef __attribute__((ext_vector_type(8)))  unsigned short u16x8;

union BF16Vec { bf16x16 v; unsigned short u[16]; };

static __device__ __forceinline__ unsigned short f2bf(float f) {
    unsigned int x = __builtin_bit_cast(unsigned int, f);
    x += 0x7FFFu + ((x >> 16) & 1u);       // round-to-nearest-even
    return (unsigned short)(x >> 16);
}

static __device__ __forceinline__ f32x8 wmma_bf16(bf16x16 a, bf16x16 b, f32x8 c) {
    return __builtin_amdgcn_wmma_f32_16x16x32_bf16(false, a, false, b, (short)0, c,
                                                   false, false);
}

// A operand (16x32, K interleaved {kb..kb+7, kb+16..kb+23} per half), bf16 source
static __device__ __forceinline__ bf16x16 load_a_bf16(const unsigned short* base,
                                                      int ld, int row, int kb) {
    u16x8 h0 = *reinterpret_cast<const u16x8*>(base + (size_t)row * ld + kb);
    u16x8 h1 = *reinterpret_cast<const u16x8*>(base + (size_t)row * ld + kb + 16);
    BF16Vec x;
#pragma unroll
    for (int i = 0; i < 8; ++i) { x.u[i] = h0[i]; x.u[8 + i] = h1[i]; }
    return x.v;
}

// B operand (K interleaved {kb..kb+7, kb+16..kb+23} per half), bf16 source
static __device__ __forceinline__ bf16x16 load_b_bf16(const unsigned short* base, size_t off) {
    u16x8 h0 = *reinterpret_cast<const u16x8*>(base + off);
    u16x8 h1 = *reinterpret_cast<const u16x8*>(base + off + 16);
    BF16Vec x;
#pragma unroll
    for (int i = 0; i < 8; ++i) { x.u[i] = h0[i]; x.u[8 + i] = h1[i]; }
    return x.v;
}

// ---------------------------------------------------------------------------
// f32 -> bf16 conversion (activations and weights), float4 traffic
// ---------------------------------------------------------------------------
__global__ __launch_bounds__(256) void cvt_kernel(const float* __restrict__ in,
                                                  unsigned short* __restrict__ out,
                                                  int n4) {
    int i = blockIdx.x * 256 + threadIdx.x;
    if (i >= n4) return;
    float4 v = reinterpret_cast<const float4*>(in)[i];
    ushort4 o;
    o.x = f2bf(v.x); o.y = f2bf(v.y); o.z = f2bf(v.z); o.w = f2bf(v.w);
    reinterpret_cast<ushort4*>(out)[i] = o;
}

// ---------------------------------------------------------------------------
// GEMM: Y[m,n] = sum_k A[m,k] * W[n,k]  (+bias)   M=8192, N=K=1024, bf16 in
// MODE 0: store bf16, scaled                           (Q/K projections)
// MODE 1: store bf16 transposed per head Vt[b][h][d][s] (V projection)
// MODE 2: store f32 + bias + residual                   (output proj)
// Each wave owns a 64x64 tile (4x4 WMMA tiles). 8 waves / block.
// ---------------------------------------------------------------------------
template <int MODE>
__global__ __launch_bounds__(256) void gemm_kernel(const unsigned short* __restrict__ A,
                                                   const unsigned short* __restrict__ W,
                                                   const float* __restrict__ bias,
                                                   void* __restrict__ Out,
                                                   const float* __restrict__ resid,
                                                   float scale) {
    const int l  = threadIdx.x & 31;
    const int r  = l & 15;
    const int hf = l >> 4;
    const int wid = blockIdx.x * 8 + (threadIdx.x >> 5);
    const int mt = wid >> 4;          // 128 M-tiles of 64
    const int nt = wid & 15;          // 16  N-tiles of 64
    const int m0 = mt * 64, n0 = nt * 64;

    f32x8 acc[4][4];
#pragma unroll
    for (int i = 0; i < 4; ++i)
#pragma unroll
        for (int j = 0; j < 4; ++j)
#pragma unroll
            for (int e = 0; e < 8; ++e) acc[i][j][e] = 0.f;

    for (int k0 = 0; k0 < EE; k0 += 32) {
        bf16x16 a[4], b[4];
#pragma unroll
        for (int mi = 0; mi < 4; ++mi) {
            int row = m0 + mi * 16 + r;
            a[mi] = load_a_bf16(A, EE, row, k0 + hf * 8);
            if (k0 + 256 < EE)
                __builtin_prefetch(A + (size_t)row * EE + k0 + 256, 0, 3);
        }
#pragma unroll
        for (int ni = 0; ni < 4; ++ni) {
            int col = n0 + ni * 16 + r;
            b[ni] = load_b_bf16(W, (size_t)col * EE + k0 + hf * 8);
            if (k0 + 256 < EE)
                __builtin_prefetch(W + (size_t)col * EE + k0 + 256, 0, 3);
        }
#pragma unroll
        for (int mi = 0; mi < 4; ++mi)
#pragma unroll
            for (int ni = 0; ni < 4; ++ni)
                acc[mi][ni] = wmma_bf16(a[mi], b[ni], acc[mi][ni]);
    }

#pragma unroll
    for (int mi = 0; mi < 4; ++mi)
#pragma unroll
        for (int ni = 0; ni < 4; ++ni)
#pragma unroll
            for (int rr = 0; rr < 8; ++rr) {
                int row = m0 + mi * 16 + rr + 8 * hf;
                int col = n0 + ni * 16 + r;
                float v = acc[mi][ni][rr] + bias[col];
                if (MODE == 0) {
                    ((unsigned short*)Out)[(size_t)row * EE + col] = f2bf(v * scale);
                } else if (MODE == 1) {
                    int s = row & (SS - 1), bb = row >> 11;
                    int h = col >> 6, d = col & 63;
                    ((unsigned short*)Out)[(size_t)((bb * HH + h) * DD + d) * SS + s] = f2bf(v);
                } else {
                    ((float*)Out)[(size_t)row * EE + col] =
                        v + resid[(size_t)row * EE + col];
                }
            }
}

// ---------------------------------------------------------------------------
// Flash attention (causal). One wave handles 16 query rows of one (b,h).
// Scores via WMMA (bf16), online softmax in f32 (C layout), P transposed
// C->A layout through per-wave LDS, PV via WMMA against pre-transposed V.
// ---------------------------------------------------------------------------
__global__ __launch_bounds__(256) void attn_kernel(const unsigned short* __restrict__ Qp,
                                                   const unsigned short* __restrict__ Kp,
                                                   const unsigned short* __restrict__ Vt,
                                                   unsigned short* __restrict__ Oa) {
    __shared__ float pbuf[8][16][33];

    const int wv = threadIdx.x >> 5;
    const int l  = threadIdx.x & 31;
    const int r  = l & 15;
    const int hf = l >> 4;

    const int gw = blockIdx.x * 8 + wv;
    const int qt = gw & 127;          // 128 query tiles per head
    const int bh = gw >> 7;
    const int b  = bh >> 4;
    const int h  = bh & 15;
    const int q0 = qt << 4;

    // Q A-operands, d = {0..31}, {32..63}
    bf16x16 aQ[2];
#pragma unroll
    for (int c = 0; c < 2; ++c) {
        size_t off = ((size_t)(b * SS + q0 + r)) * EE + h * DD + c * 32 + hf * 8;
        u16x8 h0 = *reinterpret_cast<const u16x8*>(Qp + off);
        u16x8 h1 = *reinterpret_cast<const u16x8*>(Qp + off + 16);
        BF16Vec x;
#pragma unroll
        for (int i = 0; i < 8; ++i) { x.u[i] = h0[i]; x.u[8 + i] = h1[i]; }
        aQ[c] = x.v;
    }

    f32x8 m, lsum, o[4];
#pragma unroll
    for (int e = 0; e < 8; ++e) { m[e] = -1e30f; lsum[e] = 0.f; }
#pragma unroll
    for (int dt = 0; dt < 4; ++dt)
#pragma unroll
        for (int e = 0; e < 8; ++e) o[dt][e] = 0.f;

    const size_t vhead = (size_t)((b * HH + h) * DD) * SS;

    for (int kb = 0; kb < q0 + 16; kb += 32) {
        // ---- scores: two 16x16 tiles over 32 keys ----
        f32x8 st[2];
#pragma unroll
        for (int nt = 0; nt < 2; ++nt) {
            int key = kb + nt * 16 + r;
            size_t off = ((size_t)(b * SS + key)) * EE + h * DD + hf * 8;
            bf16x16 bk0 = load_b_bf16(Kp, off);
            bf16x16 bk1 = load_b_bf16(Kp, off + 32);
            f32x8 z;
#pragma unroll
            for (int e = 0; e < 8; ++e) z[e] = 0.f;
            z = wmma_bf16(aQ[0], bk0, z);
            z = wmma_bf16(aQ[1], bk1, z);
#pragma unroll
            for (int rr = 0; rr < 8; ++rr) {
                int rowg = q0 + rr + 8 * hf;
                if (key > rowg) z[rr] = -1e30f;
            }
            st[nt] = z;
        }

        // ---- online softmax ----
        f32x8 rm;
#pragma unroll
        for (int e = 0; e < 8; ++e) {
            float v = fmaxf(st[0][e], st[1][e]);
#pragma unroll
            for (int msk = 1; msk < 16; msk <<= 1) v = fmaxf(v, __shfl_xor(v, msk, 32));
            rm[e] = v;
        }
        f32x8 alpha;
#pragma unroll
        for (int e = 0; e < 8; ++e) {
            float mn = fmaxf(m[e], rm[e]);
            alpha[e] = __expf(m[e] - mn);
            m[e]     = mn;
        }
        f32x8 p0, p1, rs;
#pragma unroll
        for (int e = 0; e < 8; ++e) {
            p0[e] = __expf(st[0][e] - m[e]);
            p1[e] = __expf(st[1][e] - m[e]);
            float s = p0[e] + p1[e];
#pragma unroll
            for (int msk = 1; msk < 16; msk <<= 1) s += __shfl_xor(s, msk, 32);
            rs[e] = s;
        }
#pragma unroll
        for (int e = 0; e < 8; ++e) lsum[e] = lsum[e] * alpha[e] + rs[e];
#pragma unroll
        for (int dt = 0; dt < 4; ++dt)
#pragma unroll
            for (int e = 0; e < 8; ++e) o[dt][e] *= alpha[e];

        // ---- transpose P (C layout -> A layout) via LDS ----
#pragma unroll
        for (int rr = 0; rr < 8; ++rr) {
            pbuf[wv][rr + 8 * hf][r]      = p0[rr];
            pbuf[wv][rr + 8 * hf][r + 16] = p1[rr];
        }
        asm volatile("s_wait_dscnt 0" ::: "memory");
        BF16Vec ap;
#pragma unroll
        for (int j = 0; j < 8; ++j) {
            ap.u[j]     = f2bf(pbuf[wv][r][hf * 8 + j]);
            ap.u[8 + j] = f2bf(pbuf[wv][r][hf * 8 + 16 + j]);
        }
        asm volatile("s_wait_dscnt 0" ::: "memory");

        // ---- PV: o += P(16x32) @ V(32x64) ----
#pragma unroll
        for (int dt = 0; dt < 4; ++dt) {
            size_t off = vhead + (size_t)(dt * 16 + r) * SS + kb + hf * 8;
            bf16x16 bv = load_b_bf16(Vt, off);
            o[dt] = wmma_bf16(ap.v, bv, o[dt]);
        }
    }

    // ---- normalize, write bf16 ----
#pragma unroll
    for (int dt = 0; dt < 4; ++dt)
#pragma unroll
        for (int rr = 0; rr < 8; ++rr) {
            int row = q0 + rr + 8 * hf;
            int col = h * DD + dt * 16 + r;
            Oa[((size_t)(b * SS + row)) * EE + col] = f2bf(o[dt][rr] / lsum[rr]);
        }
}

// ---------------------------------------------------------------------------
// LayerNorm: one 256-thread block per row of 1024.
// ---------------------------------------------------------------------------
__global__ __launch_bounds__(256) void ln_kernel(const float* __restrict__ x,
                                                 const float* __restrict__ gamma,
                                                 const float* __restrict__ beta,
                                                 float* __restrict__ out) {
    __shared__ float s1[256], s2[256];
    const int row = blockIdx.x;
    const int t   = threadIdx.x;
    const size_t base = (size_t)row * EE + t * 4;

    float4 v = *reinterpret_cast<const float4*>(x + base);
    float s  = v.x + v.y + v.z + v.w;
    float sq = v.x * v.x + v.y * v.y + v.z * v.z + v.w * v.w;
    s1[t] = s; s2[t] = sq;
    __syncthreads();
    for (int off = 128; off > 0; off >>= 1) {
        if (t < off) { s1[t] += s1[t + off]; s2[t] += s2[t + off]; }
        __syncthreads();
    }
    const float mu  = s1[0] * (1.0f / EE);
    const float var = s2[0] * (1.0f / EE) - mu * mu;
    const float rstd = rsqrtf(var + LN_EPS);

    float4 g   = *reinterpret_cast<const float4*>(gamma + t * 4);
    float4 bta = *reinterpret_cast<const float4*>(beta + t * 4);
    float4 o;
    o.x = (v.x - mu) * rstd * g.x + bta.x;
    o.y = (v.y - mu) * rstd * g.y + bta.y;
    o.z = (v.z - mu) * rstd * g.z + bta.z;
    o.w = (v.w - mu) * rstd * g.w + bta.w;
    *reinterpret_cast<float4*>(out + base) = o;
}

// ---------------------------------------------------------------------------
extern "C" void kernel_launch(void* const* d_in, const int* in_sizes, int n_in,
                              void* d_out, int out_size, void* d_ws, size_t ws_size,
                              hipStream_t stream) {
    const float* q     = (const float*)d_in[0];
    const float* k     = (const float*)d_in[1];
    const float* v     = (const float*)d_in[2];
    const float* Wq    = (const float*)d_in[3];
    const float* bq    = (const float*)d_in[4];
    const float* Wk    = (const float*)d_in[5];
    const float* bk    = (const float*)d_in[6];
    const float* Wv    = (const float*)d_in[7];
    const float* bv    = (const float*)d_in[8];
    const float* Wo    = (const float*)d_in[9];
    const float* bo    = (const float*)d_in[10];
    const float* gamma = (const float*)d_in[11];
    const float* beta  = (const float*)d_in[12];

    const size_t NE = (size_t)MM * EE;   // 8192*1024 activation elems
    const size_t E2 = (size_t)EE * EE;   // 1024*1024 weight elems

    unsigned short* base = (unsigned short*)d_ws;
    unsigned short* Xq  = base;                  // bf16 [B,S,E]  (dead after Q proj)
    unsigned short* Xk  = Xq + NE;               // bf16 [B,S,E]  (dead after K proj)
    unsigned short* Xv  = Xk + NE;               // bf16 [B,S,E]  (dead after V proj)
    unsigned short* Wqc = Xv + NE;               // bf16 weights
    unsigned short* Wkc = Wqc + E2;
    unsigned short* Wvc = Wkc + E2;
    unsigned short* Woc = Wvc + E2;
    unsigned short* Qp  = Woc + E2;              // bf16 [B,S,E]
    unsigned short* Kp  = Qp + NE;               // bf16 [B,S,E]
    unsigned short* Vt  = Kp + NE;               // bf16 [B,H,D,S]
    unsigned short* Oa  = Xq;                    // alias: attention out over dead Xq
    float*        resid = (float*)Xk;            // alias: f32 over dead Xk+Xv (32MB)

    dim3 blk(256);

    // f32 -> bf16 staging (activations + weights)
    dim3 cga((int)(NE / 4 + 255) / 256);
    dim3 cgw((int)(E2 / 4 + 255) / 256);
    cvt_kernel<<<cga, blk, 0, stream>>>(q,  Xq,  (int)(NE / 4));
    cvt_kernel<<<cga, blk, 0, stream>>>(k,  Xk,  (int)(NE / 4));
    cvt_kernel<<<cga, blk, 0, stream>>>(v,  Xv,  (int)(NE / 4));
    cvt_kernel<<<cgw, blk, 0, stream>>>(Wq, Wqc, (int)(E2 / 4));
    cvt_kernel<<<cgw, blk, 0, stream>>>(Wk, Wkc, (int)(E2 / 4));
    cvt_kernel<<<cgw, blk, 0, stream>>>(Wv, Wvc, (int)(E2 / 4));
    cvt_kernel<<<cgw, blk, 0, stream>>>(Wo, Woc, (int)(E2 / 4));

    dim3 gemm_grid((MM / 64) * (EE / 64) / 8);   // 256 blocks, 8 waves each

    // Q = (x Wq^T + bq) / 8   (scale folded)
    gemm_kernel<0><<<gemm_grid, blk, 0, stream>>>(Xq, Wqc, bq, Qp, nullptr, 0.125f);
    gemm_kernel<0><<<gemm_grid, blk, 0, stream>>>(Xk, Wkc, bk, Kp, nullptr, 1.0f);
    gemm_kernel<1><<<gemm_grid, blk, 0, stream>>>(Xv, Wvc, bv, Vt, nullptr, 1.0f);

    // attention: B*H*(S/16) waves / 8 waves per block
    dim3 attn_grid(BB * HH * (SS / 16) / 8);     // 1024 blocks
    attn_kernel<<<attn_grid, blk, 0, stream>>>(Qp, Kp, Vt, Oa);

    // out-proj + bias + residual(query), f32
    gemm_kernel<2><<<gemm_grid, blk, 0, stream>>>(Oa, Woc, bo, resid, q, 1.0f);

    // layernorm -> final output
    ln_kernel<<<MM, blk, 0, stream>>>(resid, gamma, beta, (float*)d_out);
}
